// TransformerEncoder_25804163514804
// MI455X (gfx1250) — compile-verified, matrix operand order checked by reading, not run on hardware
//
#include <hip/hip_runtime.h>
#include <hip/hip_bf16.h>

// ---------------------------------------------------------------------------
// Transformer encoder for MI455X (gfx1250, wave32, WMMA).
// L=4, B=4, T=1024, D=1024, H=16, HS=VS=64, HID=4096.
// f16 WMMA GEMMs (f32 accum), flash-attention with WMMA, f32 elementwise.
// LDS staging uses GLOBAL_LOAD_ASYNC_TO_LDS_B128 (ASYNCcnt-tracked DMA).
// ---------------------------------------------------------------------------

typedef __attribute__((ext_vector_type(16))) _Float16 v16h;
typedef __attribute__((ext_vector_type(8)))  _Float16 v8h;
typedef __attribute__((ext_vector_type(8)))  float    v8f;
typedef __attribute__((ext_vector_type(4)))  int      v4i;

union F8  { v8f v;  float f[8]; };
union H8  { v8h v;  _Float16 h[8]; };
union H16 { v16h v; v8h h2[2]; _Float16 h[16]; };

#define CB  4
#define CT  1024
#define CD  1024
#define CH  16
#define CHS 64
#define CHID 4096
#define CM  (CB*CT)
#define MASK_NEG_F (-2.3819763e+38f)

#if defined(__has_builtin)
#  if __has_builtin(__builtin_amdgcn_global_load_async_to_lds_b128)
#    define HAVE_ASYNC_LDS 1
#  endif
#endif
#ifndef HAVE_ASYNC_LDS
#  define HAVE_ASYNC_LDS 0
#endif

#define GLOBAL_AS __attribute__((address_space(1)))
#define LDS_AS    __attribute__((address_space(3)))

// 16B global -> LDS copy. Async DMA when available (ASYNCcnt), else sync.
// Builtin signature (from toolchain diagnostic): v4i* pointers.
__device__ __forceinline__ void cp_b128(const _Float16* g, _Float16* l) {
#if HAVE_ASYNC_LDS
  __builtin_amdgcn_global_load_async_to_lds_b128(
      (GLOBAL_AS v4i*)(void*)g, (LDS_AS v4i*)(void*)l, 0, 0);
#else
  *(v8h*)l = *(const v8h*)g;
#endif
}

__device__ __forceinline__ void wait_async_lds() {
#if HAVE_ASYNC_LDS
#  if __has_builtin(__builtin_amdgcn_s_wait_asynccnt)
  __builtin_amdgcn_s_wait_asynccnt(0);
#  else
  asm volatile("s_wait_asynccnt 0" ::: "memory");
#  endif
#endif
}

__device__ __forceinline__ v8f wmma_f16(v16h a, v16h b, v8f c) {
  return __builtin_amdgcn_wmma_f32_16x16x32_f16(
      false, a, false, b, (short)0, c, false, false);
}

// 16x32 f16 operand fragment from LDS (A-layout; B mirrored with lane=col).
// Lane L (row = L&15, hi = L>>4): halves 0..7 = K[8hi..], 8..15 = K[16+8hi..].
__device__ __forceinline__ v16h ldfrag(const _Float16* base, int ldm, int lane) {
  int row = lane & 15;
  int ko  = (lane >> 4) * 8;
  const _Float16* p = base + row * ldm + ko;
  H16 u;
  u.h2[0] = *(const v8h*)(p);
  u.h2[1] = *(const v8h*)(p + 16);
  return u.v;
}

// ---------------------------------------------------------------------------
// Weight cast + transpose:  W[K][N] f32 -> WT[N][K] f16  (32x32 LDS tiles)
// ---------------------------------------------------------------------------
__global__ void transpose_cast_kernel(const float* __restrict__ W,
                                      _Float16* __restrict__ WT,
                                      int K, int N) {
  __shared__ float tile[32][33];
  int n0 = blockIdx.x * 32, k0 = blockIdx.y * 32;
  int tx = threadIdx.x, ty = threadIdx.y;   // 32 x 8
#pragma unroll
  for (int j = 0; j < 32; j += 8)
    tile[ty + j][tx] = W[(size_t)(k0 + ty + j) * N + n0 + tx];
  __syncthreads();
#pragma unroll
  for (int j = 0; j < 32; j += 8)
    WT[(size_t)(n0 + ty + j) * K + k0 + tx] = (_Float16)tile[tx][ty + j];
}

// ---------------------------------------------------------------------------
// RMSNorm over D=1024 -> f16. One 256-thread block per row.
// ---------------------------------------------------------------------------
__global__ __launch_bounds__(256) void rmsnorm_kernel(const float* __restrict__ x,
                                                      const float* __restrict__ scale,
                                                      _Float16* __restrict__ y) {
  int row = blockIdx.x, tid = threadIdx.x;
  const float* xr = x + (size_t)row * CD;
  float ss = 0.f;
#pragma unroll
  for (int j = 0; j < 4; ++j) { float v = xr[tid + j * 256]; ss += v * v; }
  __shared__ float red[256];
  red[tid] = ss; __syncthreads();
  for (int s = 128; s > 0; s >>= 1) {
    if (tid < s) red[tid] += red[tid + s];
    __syncthreads();
  }
  float r = rsqrtf(red[0] * (1.f / CD) + 1e-6f);
#pragma unroll
  for (int j = 0; j < 4; ++j) {
    int c = tid + j * 256;
    y[(size_t)row * CD + c] = (_Float16)(xr[c] * r * scale[c]);
  }
}

// ---------------------------------------------------------------------------
// Tiled WMMA GEMM:  C[M][N] (f32) = A[M][K](f16) * BT[N][K](f16)^T
//                    + bias[N] (opt) + resid[M][N] (opt, may alias C)
// 256 thr = 8 waves (2x4), tile 128x128, BK=32, double-buffered async LDS.
// ---------------------------------------------------------------------------
__global__ __launch_bounds__(256) void gemm_f16f32_kernel(
    const _Float16* __restrict__ A, const _Float16* __restrict__ BT,
    const float* __restrict__ bias, const float* __restrict__ resid,
    float* __restrict__ C, int M, int N, int K) {
  __shared__ __align__(16) _Float16 sA[2][128][40];
  __shared__ __align__(16) _Float16 sB[2][128][40];

  int tid = threadIdx.x, lane = tid & 31, wid = tid >> 5;
  int m0 = blockIdx.y * 128, n0 = blockIdx.x * 128;
  int wm = wid & 1, wn = wid >> 1;

  int r0 = tid >> 2;              // 0..63
  int c0 = (tid & 3) * 8;         // 0,8,16,24 halves

  auto stage = [&](int kt, int buf) {
    size_t k0 = (size_t)kt * 32;
    cp_b128(A  + (size_t)(m0 + r0)      * K + k0 + c0, &sA[buf][r0][c0]);
    cp_b128(A  + (size_t)(m0 + r0 + 64) * K + k0 + c0, &sA[buf][r0 + 64][c0]);
    cp_b128(BT + (size_t)(n0 + r0)      * K + k0 + c0, &sB[buf][r0][c0]);
    cp_b128(BT + (size_t)(n0 + r0 + 64) * K + k0 + c0, &sB[buf][r0 + 64][c0]);
  };

  F8 acc[4][2];
#pragma unroll
  for (int mt = 0; mt < 4; ++mt)
#pragma unroll
    for (int nt = 0; nt < 2; ++nt)
#pragma unroll
      for (int r = 0; r < 8; ++r) acc[mt][nt].f[r] = 0.f;

  int nk = K >> 5;
  stage(0, 0);
  wait_async_lds();
  __syncthreads();

  for (int kt = 0; kt < nk; ++kt) {
    int cur = kt & 1;
    if (kt + 1 < nk) stage(kt + 1, cur ^ 1);   // overlap DMA with WMMA below

    v16h af[4], bf[2];
#pragma unroll
    for (int mt = 0; mt < 4; ++mt)
      af[mt] = ldfrag(&sA[cur][wm * 64 + mt * 16][0], 40, lane);
#pragma unroll
    for (int nt = 0; nt < 2; ++nt)
      bf[nt] = ldfrag(&sB[cur][wn * 32 + nt * 16][0], 40, lane);
#pragma unroll
    for (int mt = 0; mt < 4; ++mt)
#pragma unroll
      for (int nt = 0; nt < 2; ++nt)
        acc[mt][nt].v = wmma_f16(af[mt], bf[nt], acc[mt][nt].v);

    wait_async_lds();
    __syncthreads();
  }

  // Epilogue (branch-free inner loops; bias/resid tests hoisted).
  int colb = n0 + wn * 32 + (lane & 15);
  int rowb = m0 + wm * 64 + 8 * (lane >> 4);
  float bv[2] = {0.f, 0.f};
  if (bias) { bv[0] = bias[colb]; bv[1] = bias[colb + 16]; }
  if (resid) {
#pragma unroll
    for (int mt = 0; mt < 4; ++mt)
#pragma unroll
      for (int nt = 0; nt < 2; ++nt)
#pragma unroll
        for (int r = 0; r < 8; ++r) {
          size_t idx = (size_t)(rowb + mt * 16 + r) * N + colb + nt * 16;
          C[idx] = acc[mt][nt].f[r] + bv[nt] + resid[idx];
        }
  } else {
#pragma unroll
    for (int mt = 0; mt < 4; ++mt)
#pragma unroll
      for (int nt = 0; nt < 2; ++nt)
#pragma unroll
        for (int r = 0; r < 8; ++r) {
          size_t idx = (size_t)(rowb + mt * 16 + r) * N + colb + nt * 16;
          C[idx] = acc[mt][nt].f[r] + bv[nt];
        }
  }
}

// ---------------------------------------------------------------------------
// Per-head RMSNorm + RoPE. in: (B*T, H*HS) f32, out: (B*H, T, HS) f16.
// One wave per (b,t,h); lane handles dims (d, d+32).
// ---------------------------------------------------------------------------
__global__ __launch_bounds__(256) void rope_rms_kernel(const float* __restrict__ in,
                                                       const float* __restrict__ hscale,
                                                       _Float16* __restrict__ out,
                                                       float outscale) {
  int wgl  = blockIdx.x * 8 + (threadIdx.x >> 5);
  int lane = threadIdx.x & 31;
  int h = wgl & (CH - 1);
  int t = (wgl >> 4) & (CT - 1);
  int b = wgl >> 14;
  const float* p = in + ((size_t)(b * CT + t)) * (CH * CHS) + h * CHS;
  float x1 = p[lane], x2 = p[lane + 32];
  float ss = x1 * x1 + x2 * x2;
#pragma unroll
  for (int m = 16; m > 0; m >>= 1) ss += __shfl_xor(ss, m, 32);
  float r = rsqrtf(ss * (1.f / CHS) + 1e-6f);
  float y1 = x1 * r * hscale[lane];
  float y2 = x2 * r * hscale[lane + 32];
  float inv_ts = powf(10000.f, -(float)lane * (1.f / 32.f));
  float ang = (float)t * inv_ts;
  float s = sinf(ang), c = cosf(ang);
  float o1 = y1 * c - y2 * s;
  float o2 = y2 * c + y1 * s;
  _Float16* q = out + ((size_t)((b * CH + h) * CT + t)) * CHS;
  q[lane]      = (_Float16)(o1 * outscale);
  q[lane + 32] = (_Float16)(o2 * outscale);
}

// V: (B*T, H*VS) f32 -> (B*H, T, VS) f16
__global__ __launch_bounds__(256) void vperm_kernel(const float* __restrict__ vf,
                                                    _Float16* __restrict__ v16) {
  int idx = blockIdx.x * 256 + threadIdx.x;
  int d  = idx & 63;
  int h  = (idx >> 6) & 15;
  int bt = idx >> 10;
  int b = bt >> 10, t = bt & 1023;
  v16[((size_t)((b * CH + h) * CT + t)) * CHS + d] = (_Float16)vf[idx];
}

// ---------------------------------------------------------------------------
// Flash attention. Grid (T/128, B*H), block 256 (8 waves, 16 q-rows each).
// Q pre-scaled by 1/sqrt(HS). Output (B*T, H*VS) f16.
// ---------------------------------------------------------------------------
__global__ __launch_bounds__(256) void attn_kernel(const _Float16* __restrict__ Q,
                                                   const _Float16* __restrict__ Kk,
                                                   const _Float16* __restrict__ V,
                                                   const unsigned char* __restrict__ mask,
                                                   _Float16* __restrict__ O) {
  __shared__ __align__(16) _Float16 sK[128][72];       // keys x hs
  __shared__ __align__(16) _Float16 sVT[64][136];      // hs x keys
  __shared__ __align__(16) _Float16 sP[8][16][136];    // per-wave P tile
  __shared__ float sMB[128];

  int tid = threadIdx.x, lane = tid & 31, wid = tid >> 5;
  int bh = blockIdx.y, b = bh >> 4;
  int q0 = blockIdx.x * 128;

  const _Float16* qp = Q + ((size_t)bh * CT + q0 + wid * 16 + (lane & 15)) * CHS;
  int ko = (lane >> 4) * 8;
  v16h qf[2];
#pragma unroll
  for (int kc = 0; kc < 2; ++kc) {
    H16 u;
    u.h2[0] = *(const v8h*)(qp + kc * 32 + ko);
    u.h2[1] = *(const v8h*)(qp + kc * 32 + ko + 16);
    qf[kc] = u.v;
  }

  float m[8], l[8], alpha[8];
  F8 o[4];
#pragma unroll
  for (int r = 0; r < 8; ++r) { m[r] = -3.0e38f; l[r] = 0.f; }
#pragma unroll
  for (int ot = 0; ot < 4; ++ot)
#pragma unroll
    for (int r = 0; r < 8; ++r) o[ot].f[r] = 0.f;

  for (int j = 0; j < CT / 128; ++j) {
    int kv0 = j * 128;
    // K tile: direct async DMA to LDS
#pragma unroll
    for (int c = 0; c < 4; ++c) {
      int chunk = tid + c * 256, row = chunk >> 3, cc = (chunk & 7) * 8;
      cp_b128(Kk + ((size_t)bh * CT + kv0 + row) * CHS + cc, &sK[row][cc]);
    }
    // V tile transposed (needs lane shuffle through VGPRs)
#pragma unroll
    for (int c = 0; c < 4; ++c) {
      int chunk = tid + c * 256, row = chunk >> 3, cc = (chunk & 7) * 8;
      H8 vv; vv.v = *(const v8h*)(V + ((size_t)bh * CT + kv0 + row) * CHS + cc);
#pragma unroll
      for (int e = 0; e < 8; ++e) sVT[cc + e][row] = vv.h[e];
    }
    if (tid < 128) sMB[tid] = mask[b * CT + kv0 + tid] ? 0.f : MASK_NEG_F;
    wait_async_lds();
    __syncthreads();

    // S = Q K^T (16 x 128 per wave)
    F8 s[8];
#pragma unroll
    for (int nt = 0; nt < 8; ++nt) {
#pragma unroll
      for (int r = 0; r < 8; ++r) s[nt].f[r] = 0.f;
#pragma unroll
      for (int kc = 0; kc < 2; ++kc) {
        v16h bf = ldfrag(&sK[nt * 16][kc * 32], 72, lane);
        s[nt].v = wmma_f16(qf[kc], bf, s[nt].v);
      }
    }

    // Mask bias + online-softmax stats (rows live on 16-lane halves)
    float tmax[8];
#pragma unroll
    for (int r = 0; r < 8; ++r) tmax[r] = -3.0e38f;
#pragma unroll
    for (int nt = 0; nt < 8; ++nt) {
      float mb = sMB[nt * 16 + (lane & 15)];
#pragma unroll
      for (int r = 0; r < 8; ++r) {
        s[nt].f[r] += mb;
        tmax[r] = fmaxf(tmax[r], s[nt].f[r]);
      }
    }
#pragma unroll
    for (int r = 0; r < 8; ++r) {
#pragma unroll
      for (int xm = 8; xm > 0; xm >>= 1)
        tmax[r] = fmaxf(tmax[r], __shfl_xor(tmax[r], xm, 16));
      float mn = fmaxf(m[r], tmax[r]);
      alpha[r] = __expf(m[r] - mn);
      m[r] = mn;
    }
    float rsum[8];
#pragma unroll
    for (int r = 0; r < 8; ++r) rsum[r] = 0.f;
#pragma unroll
    for (int nt = 0; nt < 8; ++nt)
#pragma unroll
      for (int r = 0; r < 8; ++r) {
        float p = __expf(s[nt].f[r] - m[r]);
        s[nt].f[r] = p;
        rsum[r] += p;
      }
#pragma unroll
    for (int r = 0; r < 8; ++r) {
#pragma unroll
      for (int xm = 8; xm > 0; xm >>= 1) rsum[r] += __shfl_xor(rsum[r], xm, 16);
      l[r] = l[r] * alpha[r] + rsum[r];
    }

    // P: C-layout -> row-major LDS (A-fragment reload)
#pragma unroll
    for (int nt = 0; nt < 8; ++nt)
#pragma unroll
      for (int r = 0; r < 8; ++r)
        sP[wid][r + 8 * (lane >> 4)][nt * 16 + (lane & 15)] = (_Float16)s[nt].f[r];
    asm volatile("s_wait_dscnt 0" ::: "memory");   // wave-private RAW on sP

#pragma unroll
    for (int ot = 0; ot < 4; ++ot)
#pragma unroll
      for (int r = 0; r < 8; ++r) o[ot].f[r] *= alpha[r];
#pragma unroll
    for (int kc = 0; kc < 4; ++kc) {
      v16h af = ldfrag(&sP[wid][0][kc * 32], 136, lane);
#pragma unroll
      for (int ot = 0; ot < 4; ++ot) {
        v16h bf = ldfrag(&sVT[ot * 16][kc * 32], 136, lane);
        o[ot].v = wmma_f16(af, bf, o[ot].v);
      }
    }
    __syncthreads();
  }

  // Normalize + write (B*T, H*VS) f16
#pragma unroll
  for (int ot = 0; ot < 4; ++ot)
#pragma unroll
    for (int r = 0; r < 8; ++r) {
      int trow = q0 + wid * 16 + r + 8 * (lane >> 4);
      unsigned char qm = mask[b * CT + trow];
      float denom = l[r];
      float val = (qm && denom > 0.f) ? o[ot].f[r] / denom : 0.f;
      O[((size_t)(b * CT) + trow) * (CH * CHS) + (bh & 15) * CHS + ot * 16 +
        (lane & 15)] = (_Float16)val;
    }
}

// gate = gelu_tanh(g) * u -> f16
__global__ __launch_bounds__(256) void gelu_mul_kernel(const float* __restrict__ g32,
                                                       const float* __restrict__ u32,
                                                       _Float16* __restrict__ h16) {
  int i = blockIdx.x * 256 + threadIdx.x;
  float g = g32[i], u = u32[i];
  float t = tanhf(0.7978845608f * (g + 0.044715f * g * g * g));
  h16[i] = (_Float16)(0.5f * g * (1.f + t) * u);
}

// zero rows where mask == 0
__global__ __launch_bounds__(256) void posmask_kernel(const unsigned char* __restrict__ mask,
                                                      float* __restrict__ x) {
  int row = blockIdx.x, tid = threadIdx.x;
  if (!mask[row]) {
#pragma unroll
    for (int j = 0; j < 4; ++j) x[(size_t)row * CD + tid + j * 256] = 0.f;
  }
}

// ---------------------------------------------------------------------------
extern "C" void kernel_launch(void* const* d_in, const int* in_sizes, int n_in,
                              void* d_out, int out_size, void* d_ws, size_t ws_size,
                              hipStream_t stream) {
  (void)in_sizes; (void)n_in; (void)out_size; (void)ws_size;
  const float* x           = (const float*)d_in[0];
  const unsigned char* msk = (const unsigned char*)d_in[1];
  const float* Wq = (const float*)d_in[2];
  const float* bq = (const float*)d_in[3];
  const float* Wk = (const float*)d_in[4];
  const float* bk = (const float*)d_in[5];
  const float* Wv = (const float*)d_in[6];
  const float* bv = (const float*)d_in[7];
  const float* Wo = (const float*)d_in[8];
  const float* bo = (const float*)d_in[9];
  const float* q_scale   = (const float*)d_in[10];
  const float* k_scale   = (const float*)d_in[11];
  const float* ln1_scale = (const float*)d_in[12];
  const float* ln2_scale = (const float*)d_in[13];
  const float* Wg = (const float*)d_in[14];
  const float* W1 = (const float*)d_in[15];
  const float* W2 = (const float*)d_in[16];
  float* out = (float*)d_out;

  const size_t NROW = CM;
  const size_t E4M  = (size_t)CM * CD;

  char* ws = (char*)d_ws;
  size_t off = 0;
  auto alloc = [&](size_t bytes) -> void* {
    void* p = ws + off;
    off += (bytes + 255) & ~(size_t)255;
    return p;
  };
  _Float16* xln16  = (_Float16*)alloc(E4M * 2);
  float*    qf     = (float*)alloc(E4M * 4);
  float*    kf     = (float*)alloc(E4M * 4);
  float*    vf     = (float*)alloc(E4M * 4);
  _Float16* q16    = (_Float16*)alloc(E4M * 2);
  _Float16* k16    = (_Float16*)alloc(E4M * 2);
  _Float16* v16    = (_Float16*)alloc(E4M * 2);
  _Float16* attn16 = (_Float16*)alloc(E4M * 2);
  float*    g32    = (float*)alloc((size_t)CM * CHID * 4);
  float*    u32    = (float*)alloc((size_t)CM * CHID * 4);
  _Float16* h16    = (_Float16*)alloc((size_t)CM * CHID * 2);
  _Float16* wtq    = (_Float16*)alloc((size_t)CD * CD * 2);
  _Float16* wtk    = (_Float16*)alloc((size_t)CD * CD * 2);
  _Float16* wtv    = (_Float16*)alloc((size_t)CD * CD * 2);
  _Float16* wto    = (_Float16*)alloc((size_t)CD * CD * 2);
  _Float16* wtg    = (_Float16*)alloc((size_t)CD * CHID * 2);
  _Float16* wt1    = (_Float16*)alloc((size_t)CD * CHID * 2);
  _Float16* wt2    = (_Float16*)alloc((size_t)CHID * CD * 2);

  (void)hipMemcpyAsync(out, x, E4M * 4, hipMemcpyDeviceToDevice, stream);

  const float inv_sqrt_hs = 0.125f;
  dim3 blk256(256), blkTC(32, 8);
  dim3 gGemmN1(CD / 128, CM / 128);
  dim3 gGemmN4(CHID / 128, CM / 128);
  dim3 gTC_1k(CD / 32, CD / 32);
  dim3 gTC_g(CHID / 32, CD / 32);
  dim3 gTC_2(CD / 32, CHID / 32);
  dim3 gAttn(CT / 128, CB * CH);

  for (int lyr = 0; lyr < 4; ++lyr) {
    const float* lWq = Wq + (size_t)lyr * CD * CD;
    const float* lWk = Wk + (size_t)lyr * CD * CD;
    const float* lWv = Wv + (size_t)lyr * CD * CD;
    const float* lWo = Wo + (size_t)lyr * CD * CD;
    const float* lWg = Wg + (size_t)lyr * CD * CHID;
    const float* lW1 = W1 + (size_t)lyr * CD * CHID;
    const float* lW2 = W2 + (size_t)lyr * CHID * CD;

    rmsnorm_kernel<<<NROW, blk256, 0, stream>>>(out, ln1_scale + lyr * CD, xln16);

    transpose_cast_kernel<<<gTC_1k, blkTC, 0, stream>>>(lWq, wtq, CD, CD);
    transpose_cast_kernel<<<gTC_1k, blkTC, 0, stream>>>(lWk, wtk, CD, CD);
    transpose_cast_kernel<<<gTC_1k, blkTC, 0, stream>>>(lWv, wtv, CD, CD);
    gemm_f16f32_kernel<<<gGemmN1, blk256, 0, stream>>>(xln16, wtq, bq + lyr * CD, nullptr, qf, CM, CD, CD);
    gemm_f16f32_kernel<<<gGemmN1, blk256, 0, stream>>>(xln16, wtk, bk + lyr * CD, nullptr, kf, CM, CD, CD);
    gemm_f16f32_kernel<<<gGemmN1, blk256, 0, stream>>>(xln16, wtv, bv + lyr * CD, nullptr, vf, CM, CD, CD);

    rope_rms_kernel<<<(CB * CT * CH) / 8, blk256, 0, stream>>>(qf, q_scale + lyr * CHS, q16, inv_sqrt_hs);
    rope_rms_kernel<<<(CB * CT * CH) / 8, blk256, 0, stream>>>(kf, k_scale + lyr * CHS, k16, 1.0f);
    vperm_kernel<<<E4M / 256, blk256, 0, stream>>>(vf, v16);

    attn_kernel<<<gAttn, blk256, 0, stream>>>(q16, k16, v16, msk, attn16);

    transpose_cast_kernel<<<gTC_1k, blkTC, 0, stream>>>(lWo, wto, CD, CD);
    gemm_f16f32_kernel<<<gGemmN1, blk256, 0, stream>>>(attn16, wto, bo + lyr * CD, out, out, CM, CD, CD);

    rmsnorm_kernel<<<NROW, blk256, 0, stream>>>(out, ln2_scale + lyr * CD, xln16);
    transpose_cast_kernel<<<gTC_g, blkTC, 0, stream>>>(lWg, wtg, CD, CHID);
    gemm_f16f32_kernel<<<gGemmN4, blk256, 0, stream>>>(xln16, wtg, nullptr, nullptr, g32, CM, CHID, CD);
    transpose_cast_kernel<<<gTC_g, blkTC, 0, stream>>>(lW1, wt1, CD, CHID);
    gemm_f16f32_kernel<<<gGemmN4, blk256, 0, stream>>>(xln16, wt1, nullptr, nullptr, u32, CM, CHID, CD);
    gelu_mul_kernel<<<((size_t)CM * CHID) / 256, blk256, 0, stream>>>(g32, u32, h16);
    transpose_cast_kernel<<<gTC_2, blkTC, 0, stream>>>(lW2, wt2, CHID, CD);
    gemm_f16f32_kernel<<<gGemmN1, blk256, 0, stream>>>(h16, wt2, nullptr, out, out, CM, CD, CHID);

    posmask_kernel<<<NROW, blk256, 0, stream>>>(msk, out);
  }
}
